// TAMKOT_3676492005948
// MI455X (gfx1250) — matrix operand order repaired and checked by reading, not compiled, not run on hardware
//
#include <hip/hip_runtime.h>

#define Bn   256
#define Sn   500
#define Hn   256
#define G4H  1024      // 4*H
#define EQn  128
#define Tn   498       // S-2

typedef __attribute__((ext_vector_type(16))) __bf16       v16bf;
typedef __attribute__((ext_vector_type(8)))  float        v8f;
typedef __attribute__((ext_vector_type(4)))  unsigned int u32x4;

static __device__ __forceinline__ v8f wmma_bf16(v16bf a, v16bf b, v8f c) {
    // D = A(16x32 bf16) * B(32x16 bf16) + C(16x16 f32)
    return __builtin_amdgcn_wmma_f32_16x16x32_bf16(
        /*neg_a=*/false, a, /*neg_b=*/false, b,
        /*c_mod=*/(short)0, c, /*reuse_a=*/false, /*reuse_b=*/false);
}

static __device__ __forceinline__ v16bf zero16() {
    v16bf z;
#pragma unroll
    for (int i = 0; i < 16; ++i) z[i] = (__bf16)0.0f;
    return z;
}

// Packed B fragment (32K x 16N tile): 32 lanes x 16 halves, lane-contiguous.
static __device__ __forceinline__ v16bf load_bfrag(const __bf16* p, int lane) {
    v16bf v;
    u32x4* pv = reinterpret_cast<u32x4*>(&v);
    const u32x4* s = reinterpret_cast<const u32x4*>(p + lane * 16);
    pv[0] = s[0];
    pv[1] = s[1];
    return v;
}

// A fragment (16M x 32K) from row-major bf16, row stride ldr.
// Lanes 0-15: row=lane, K = k0+0..7 and k0+16..23.  Lanes 16-31: K = +8.. and +24..
static __device__ __forceinline__ v16bf load_afrag_bf16(const __bf16* base, int ldr,
                                                        int row, int k0, int lane) {
    const __bf16* p = base + (size_t)row * ldr + k0 + ((lane >> 4) << 3);
    v16bf v;
    u32x4* pv = reinterpret_cast<u32x4*>(&v);
    pv[0] = *reinterpret_cast<const u32x4*>(p);
    pv[1] = *reinterpret_cast<const u32x4*>(p + 16);
    return v;
}

// 8 contiguous fp32 -> 8 bf16 (32B-aligned source).
static __device__ __forceinline__ void cvt8(const float* p, __bf16* dst) {
    v8f f = *reinterpret_cast<const v8f*>(p);
#pragma unroll
    for (int i = 0; i < 8; ++i) dst[i] = (__bf16)f[i];
}

static __device__ __forceinline__ float sigmoidf_(float x) {
    return 1.0f / (1.0f + __expf(-x));
}
static __device__ __forceinline__ float tanhf_(float x) {
    float e = __expf(2.0f * x);
    return (e - 1.0f) / (e + 1.0f);
}

// ---------------------------------------------------------------------------
// Pack a K x N fp32 row-major matrix into bf16 WMMA B-fragment layout:
// tile (kt,nt) of 32x16; within tile: lane = n + 16*((k>>3)&1), slot = (k&7) + 8*(k>>4).
// ---------------------------------------------------------------------------
__global__ void tamkot_pack_kernel(const float* __restrict__ src,
                                   __bf16* __restrict__ dst, int K, int N) {
    int e = blockIdx.x * blockDim.x + threadIdx.x;
    if (e >= K * N) return;
    int k = e / N, n = e % N;
    int kt = k >> 5, nt = n >> 4;
    int kk = k & 31, nn = n & 15;
    int lane = nn + ((kk & 8) ? 16 : 0);
    int slot = (kk & 7) | ((kk >> 4) << 3);
    dst[(size_t)(kt * (N >> 4) + nt) * 512 + lane * 16 + slot] = (__bf16)src[e];
}

// ---------------------------------------------------------------------------
// Phase A: x[b,s,:] = (1-d)*(concat(qe,ae)@WQ + bQ) + d*(le@WL + bL), bf16 out.
// One wave per block; block tile = 16 rows x 64 cols (4 WMMA N-tiles).
// Row masking of A fragments implements the d-select exactly.
// ---------------------------------------------------------------------------
__global__ __launch_bounds__(32)
void tamkot_input_proj(const int* __restrict__ q_data, const int* __restrict__ a_data,
                       const int* __restrict__ l_data, const int* __restrict__ d_data,
                       const float* __restrict__ q_embed, const float* __restrict__ a_embed,
                       const float* __restrict__ l_embed,
                       const __bf16* __restrict__ WQp, const __bf16* __restrict__ WLp,
                       const float* __restrict__ bQ, const float* __restrict__ bL,
                       __bf16* __restrict__ x) {
    const int lane  = threadIdx.x;
    const int mBase = blockIdx.x * 16;
    const int nGrp  = blockIdx.y;          // 0..15, 64 columns each

    const int rowA = mBase + (lane & 15);
    const int qid = q_data[rowA];
    const int aid = a_data[rowA];
    const int lid = l_data[rowA];
    const int dA  = d_data[rowA];

    const bool useQ = (dA == 0);
    const bool useL = (dA != 0);
    const bool anyQ = (__ballot(useQ) != 0ull);
    const bool anyL = (__ballot(useL) != 0ull);

    v8f acc[4];
#pragma unroll
    for (int t = 0; t < 4; ++t)
#pragma unroll
        for (int r = 0; r < 8; ++r) acc[t][r] = 0.0f;

    const float* qrow = q_embed + (size_t)qid * EQn;
    const float* arow = a_embed + (size_t)aid * EQn;
    const float* lrow = l_embed + (size_t)lid * EQn;

    if (anyQ) {
#pragma unroll
        for (int kk = 0; kk < 8; ++kk) {            // K = 256 over [qe | ae]
            v16bf af;
            __bf16* ap = reinterpret_cast<__bf16*>(&af);
            int k0 = kk * 32 + ((lane >> 4) << 3);
            int k1 = k0 + 16;
            const float* s0 = (k0 < EQn) ? (qrow + k0) : (arow + (k0 - EQn));
            const float* s1 = (k1 < EQn) ? (qrow + k1) : (arow + (k1 - EQn));
            cvt8(s0, ap);
            cvt8(s1, ap + 8);
            if (!useQ) af = zero16();
#pragma unroll
            for (int t = 0; t < 4; ++t) {
                v16bf bf = load_bfrag(WQp + (size_t)(kk * 64 + (nGrp * 4 + t)) * 512, lane);
                acc[t] = wmma_bf16(af, bf, acc[t]);
            }
        }
    }
    if (anyL) {
#pragma unroll
        for (int kk = 0; kk < 4; ++kk) {            // K = 128 over le
            v16bf af;
            __bf16* ap = reinterpret_cast<__bf16*>(&af);
            int k0 = kk * 32 + ((lane >> 4) << 3);
            cvt8(lrow + k0, ap);
            cvt8(lrow + k0 + 16, ap + 8);
            if (!useL) af = zero16();
#pragma unroll
            for (int t = 0; t < 4; ++t) {
                v16bf bf = load_bfrag(WLp + (size_t)(kk * 64 + (nGrp * 4 + t)) * 512, lane);
                acc[t] = wmma_bf16(af, bf, acc[t]);
            }
        }
    }

    // Epilogue: add per-row selected bias, store bf16.
#pragma unroll
    for (int r = 0; r < 8; ++r) {
        int row = mBase + r + ((lane >> 4) << 3);
        int dd  = d_data[row];
#pragma unroll
        for (int t = 0; t < 4; ++t) {
            int col = nGrp * 64 + t * 16 + (lane & 15);
            float bias = dd ? bL[col] : bQ[col];
            x[(size_t)row * G4H + col] = (__bf16)(acc[t][r] + bias);
        }
    }
}

// ---------------------------------------------------------------------------
// Phase B cell: z = x_t + h @ Wh_sel(dt,dt1); LSTM gates; m/h update.
// One wave per block; block = 16 batch rows x 16 hidden cols across 4 gates.
// combo c = dt + 2*dt1 selects Wh via per-lane A-row masking; ballot skips
// combos absent from the 16-row group.
// ---------------------------------------------------------------------------
__global__ __launch_bounds__(32)
void tamkot_cell(const __bf16* __restrict__ x,        // (B*S, 4H) bf16
                 const int* __restrict__ d_data,      // (B,S)
                 const __bf16* __restrict__ Whp,      // packed, 4 x (256x1024)
                 const __bf16* __restrict__ h_prev,   // (B,H) bf16
                 __bf16* __restrict__ h_next,         // (B,H) bf16
                 float* __restrict__ m,               // (B,H) f32
                 int tau) {
    const int lane  = threadIdx.x;
    const int mTile = blockIdx.x;     // batch tile 0..15
    const int jTile = blockIdx.y;     // hidden tile 0..15
    const int s = tau + 1;

    const int rowA  = mTile * 16 + (lane & 15);
    const int dt    = d_data[rowA * Sn + s];
    const int dt1   = d_data[rowA * Sn + s - 1];
    const int combo = dt + 2 * dt1;   // 0:QQ 1:QL 2:LQ 3:LL

    bool present[4];
#pragma unroll
    for (int c = 0; c < 4; ++c) present[c] = (__ballot(combo == c) != 0ull);

    // Init accumulators (C layout) from x_t.
    v8f acc[4];
#pragma unroll
    for (int r = 0; r < 8; ++r) {
        int row = mTile * 16 + r + ((lane >> 4) << 3);
        const __bf16* xr = x + ((size_t)row * Sn + s) * G4H;
#pragma unroll
        for (int g = 0; g < 4; ++g)
            acc[g][r] = (float)xr[g * Hn + jTile * 16 + (lane & 15)];
    }

#pragma unroll
    for (int kk = 0; kk < 8; ++kk) {              // K = H = 256
        v16bf hf = load_afrag_bf16(h_prev, Hn, rowA, kk * 32, lane);
#pragma unroll
        for (int c = 0; c < 4; ++c) {
            if (!present[c]) continue;
            v16bf af = (combo == c) ? hf : zero16();
            const __bf16* wp = Whp + (size_t)c * (256 * 1024 / 2) * 2   // 256x1024 halves
                                   + (size_t)kk * 64 * 512;
#pragma unroll
            for (int g = 0; g < 4; ++g) {
                v16bf bf = load_bfrag(wp + (size_t)(g * 16 + jTile) * 512, lane);
                acc[g] = wmma_bf16(af, bf, acc[g]);
            }
        }
    }

    // Gates + state update.
#pragma unroll
    for (int r = 0; r < 8; ++r) {
        int row = mTile * 16 + r + ((lane >> 4) << 3);
        int col = jTile * 16 + (lane & 15);
        float iv = sigmoidf_(acc[0][r]);
        float gv = tanhf_(acc[1][r]);
        float fv = sigmoidf_(acc[2][r]);
        float ov = sigmoidf_(acc[3][r]);
        float mOld = m[(size_t)row * Hn + col];
        float mNew = fv * mOld + iv * gv;
        m[(size_t)row * Hn + col] = mNew;
        h_next[(size_t)row * Hn + col] = (__bf16)(ov * tanhf_(mNew));
    }
}

// ---------------------------------------------------------------------------
// Prediction head: pred[b,tau] = sigmoid([h | q_{tau+2}] @ W_Qh + b_Qh).
// One wave per batch row, lane-parallel dot + xor-shuffle reduce (wave32).
// ---------------------------------------------------------------------------
__global__ __launch_bounds__(256)
void tamkot_pred(const __bf16* __restrict__ h,        // (B,H) bf16 (h_next)
                 const int* __restrict__ q_data,
                 const float* __restrict__ q_embed,
                 const float* __restrict__ W_Qh,      // (H+EQ)
                 const float* __restrict__ b_Qh,
                 float* __restrict__ out, int tau) {
    const int lane = threadIdx.x & 31;
    const int wave = threadIdx.x >> 5;
    const int row  = blockIdx.x * 8 + wave;
    const int qid  = q_data[row * Sn + tau + 2];
    const float* qrow = q_embed + (size_t)qid * EQn;

    float sum = 0.0f;
#pragma unroll
    for (int k = lane; k < Hn; k += 32)
        sum += (float)h[(size_t)row * Hn + k] * W_Qh[k];
#pragma unroll
    for (int k = lane; k < EQn; k += 32)
        sum += qrow[k] * W_Qh[Hn + k];
#pragma unroll
    for (int off = 16; off > 0; off >>= 1)
        sum += __shfl_xor(sum, off, 32);
    if (lane == 0)
        out[(size_t)row * Tn + tau] = sigmoidf_(sum + b_Qh[0]);
}

// ---------------------------------------------------------------------------
extern "C" void kernel_launch(void* const* d_in, const int* in_sizes, int n_in,
                              void* d_out, int out_size, void* d_ws, size_t ws_size,
                              hipStream_t stream) {
    (void)in_sizes; (void)n_in; (void)out_size; (void)ws_size;

    const int*   q_data  = (const int*)d_in[0];
    const int*   a_data  = (const int*)d_in[1];
    const int*   l_data  = (const int*)d_in[2];
    const int*   d_data  = (const int*)d_in[3];
    const float* q_embed = (const float*)d_in[4];
    const float* a_embed = (const float*)d_in[5];
    const float* l_embed = (const float*)d_in[6];
    const float* WQ      = (const float*)d_in[7];
    const float* bQ      = (const float*)d_in[8];
    const float* WL      = (const float*)d_in[9];
    const float* bL      = (const float*)d_in[10];
    const float* WhQQ    = (const float*)d_in[11];
    const float* WhLL    = (const float*)d_in[12];
    const float* WhQL    = (const float*)d_in[13];
    const float* WhLQ    = (const float*)d_in[14];
    const float* W_Qh    = (const float*)d_in[15];
    const float* b_Qh    = (const float*)d_in[16];
    float*       out     = (float*)d_out;

    // Workspace carve-up (256B aligned).
    char*  ws  = (char*)d_ws;
    size_t off = 0;
    auto carve = [&](size_t bytes) -> char* {
        char* p = ws + off;
        off += (bytes + 255) & ~(size_t)255;
        return p;
    };
    __bf16* WQp = (__bf16*)carve((size_t)256 * 1024 * 2);
    __bf16* WLp = (__bf16*)carve((size_t)128 * 1024 * 2);
    __bf16* Whp = (__bf16*)carve((size_t)4 * 256 * 1024 * 2);
    __bf16* x   = (__bf16*)carve((size_t)Bn * Sn * G4H * 2);   // ~256 MB
    __bf16* h0  = (__bf16*)carve((size_t)Bn * Hn * 2);
    __bf16* h1  = (__bf16*)carve((size_t)Bn * Hn * 2);
    float*  mst = (float*)carve((size_t)Bn * Hn * 4);

    // Pack weights into WMMA B-fragment bf16 layout.
    auto pack = [&](const float* src, __bf16* dst, int K, int N) {
        int tot = K * N;
        tamkot_pack_kernel<<<(tot + 255) / 256, 256, 0, stream>>>(src, dst, K, N);
    };
    pack(WQ,   WQp,                 256, 1024);
    pack(WL,   WLp,                 128, 1024);
    pack(WhQQ, Whp + 0 * 262144,    256, 1024);   // combo 0: dt=0,dt1=0
    pack(WhQL, Whp + 1 * 262144,    256, 1024);   // combo 1: dt=1,dt1=0
    pack(WhLQ, Whp + 2 * 262144,    256, 1024);   // combo 2: dt=0,dt1=1
    pack(WhLL, Whp + 3 * 262144,    256, 1024);   // combo 3: dt=1,dt1=1

    hipMemsetAsync(h0,  0, (size_t)Bn * Hn * sizeof(__bf16), stream);
    hipMemsetAsync(mst, 0, (size_t)Bn * Hn * sizeof(float),  stream);

    // Phase A: big parallel input-projection GEMM (128000 x 1024, K=256/128).
    tamkot_input_proj<<<dim3((Bn * Sn) / 16, 16), 32, 0, stream>>>(
        q_data, a_data, l_data, d_data, q_embed, a_embed, l_embed,
        WQp, WLp, bQ, bL, x);

    // Phase B: serial scan, one cell + one pred kernel per timestep.
    __bf16* hp = h0;
    __bf16* hn = h1;
    for (int tau = 0; tau < Tn; ++tau) {
        tamkot_cell<<<dim3(16, 16), 32, 0, stream>>>(x, d_data, Whp, hp, hn, mst, tau);
        tamkot_pred<<<dim3(32), 256, 0, stream>>>(hn, q_data, q_embed, W_Qh, b_Qh, out, tau);
        __bf16* t = hp; hp = hn; hn = t;
    }
}
